// Decoder_8572754723322
// MI455X (gfx1250) — matrix operand order repaired and checked by reading, not compiled, not run on hardware
//
#include <hip/hip_runtime.h>
#include <hip/hip_bf16.h>
#include <math.h>

typedef __bf16 bf16;
typedef __attribute__((ext_vector_type(16))) __bf16 v16bf;
typedef __attribute__((ext_vector_type(8)))  __bf16 v8bf;
typedef __attribute__((ext_vector_type(8)))  float  v8f;

#define BQ    32
#define TENC  512
#define DENC  512
#define NMEL  80
#define TOUT  500
#define ARNN  1024
#define PREN  256
#define ADIM  128
#define LF    32
#define LK    31
#define NG    4096
#define KATT  1792   // [prenet 256 | ctx 512 | ah 1024]
#define KDEC  2560   // [ah 1024 | ctx 512 | dh 1024]

__device__ __forceinline__ float sigmoidf_(float x) { return 1.0f / (1.0f + __expf(-x)); }

// ---------------------------------------------------------------------------
// Setup: pack [W_ih | W_hh] fp32 -> bf16 contiguous (N x (Kih+Khh)) row-major
// ---------------------------------------------------------------------------
__global__ void pack_w_kernel(const float* __restrict__ ih, const float* __restrict__ hh,
                              bf16* __restrict__ out, int Kih, int Khh) {
    const int K = Kih + Khh;
    const long total = (long)NG * K;
    for (long idx = (long)blockIdx.x * blockDim.x + threadIdx.x; idx < total;
         idx += (long)gridDim.x * blockDim.x) {
        int n = (int)(idx / K), k = (int)(idx % K);
        float v = (k < Kih) ? ih[(long)n * Kih + k] : hh[(long)n * Khh + (k - Kih)];
        out[idx] = (bf16)v;
    }
}

__global__ void zero_u32_kernel(unsigned* __restrict__ p, int n) {
    int i = blockIdx.x * blockDim.x + threadIdx.x;
    if (i < n) p[i] = 0u;
}

// ---------------------------------------------------------------------------
// Setup: prenet for all steps. grid (TOUT, B), block 256.
// pren_bf16[(s*B + b)*PREN + j] = relu(relu(x @ w1^T) @ w2^T), x = frame s-1 (s=0 -> go)
// ---------------------------------------------------------------------------
__global__ void prenet_kernel(const float* __restrict__ dec_inputs,  // (B, NMEL, TOUT)
                              const float* __restrict__ w1,          // (PREN, NMEL)
                              const float* __restrict__ w2,          // (PREN, PREN)
                              bf16* __restrict__ pren) {
    __shared__ float s_x[NMEL];
    __shared__ float s_h1[PREN];
    const int s = blockIdx.x, b = blockIdx.y, tid = threadIdx.x;
    if (tid < NMEL)
        s_x[tid] = (s == 0) ? 0.0f : dec_inputs[(long)b * NMEL * TOUT + tid * TOUT + (s - 1)];
    __syncthreads();
    float a = 0.0f;
    #pragma unroll 4
    for (int k = 0; k < NMEL; ++k) a += w1[tid * NMEL + k] * s_x[k];
    s_h1[tid] = fmaxf(a, 0.0f);
    __syncthreads();
    float o = 0.0f;
    #pragma unroll 4
    for (int k = 0; k < PREN; ++k) o += w2[tid * PREN + k] * s_h1[k];
    pren[((long)s * BQ + b) * PREN + tid] = (bf16)fmaxf(o, 0.0f);
}

// ---------------------------------------------------------------------------
// Setup: processed_memory = memory @ memory_W^T  -> (B, T, ADIM) fp32
// grid (TENC, B), block 128.
// ---------------------------------------------------------------------------
__global__ void pmem_kernel(const float* __restrict__ memory,   // (B, T, DENC)
                            const float* __restrict__ mW,       // (ADIM, DENC)
                            float* __restrict__ pmem) {
    __shared__ float s_m[DENC];
    const int t = blockIdx.x, b = blockIdx.y, tid = threadIdx.x;
    const float* mrow = memory + ((long)b * TENC + t) * DENC;
    for (int i = tid; i < DENC; i += 128) s_m[i] = mrow[i];
    __syncthreads();
    float acc = 0.0f;
    #pragma unroll 4
    for (int k = 0; k < DENC; ++k) acc += mW[tid * DENC + k] * s_m[k];
    pmem[((long)b * TENC + t) * ADIM + tid] = acc;
}

// ---------------------------------------------------------------------------
// Per-step: copy precomputed prenet frame into X_att[:, 0:256]
// ---------------------------------------------------------------------------
__global__ void copy_pren_kernel(const bf16* __restrict__ pren_s,  // (B, PREN) for this step
                                 bf16* __restrict__ xatt) {        // (B, KATT)
    int idx = blockIdx.x * blockDim.x + threadIdx.x;  // 0 .. B*PREN-1
    int b = idx >> 8, j = idx & 255;
    xatt[(long)b * KATT + j] = pren_s[idx];
}

// ---------------------------------------------------------------------------
// Core per-step GEMM: gates(32 x 4096) = A(32 x K, bf16) x W^T(K x 4096, bf16 as N x K)
// One wave per 16x16 D tile: 2 M-tiles x 256 N-tiles = 512 tiles.
// Grid 64 blocks x 256 threads (8 waves). K multiple of 32.
// VGPR layouts per CDNA5 ISA 7.12.2 (16-bit A 16x32, B 32x16, f32 C/D 16x16).
// ---------------------------------------------------------------------------
__global__ void gemm_gates_kernel(const bf16* __restrict__ A,
                                  const bf16* __restrict__ W,
                                  float* __restrict__ out, int K) {
    const int wave = threadIdx.x >> 5;
    const int lane = threadIdx.x & 31;
    const int tile = blockIdx.x * 8 + wave;     // 0..511
    const int nt = tile >> 1, mt = tile & 1;
    const int l16 = lane & 15, g = lane >> 4;
    const bf16* __restrict__ arow = A + (long)(mt * 16 + l16) * K;
    const bf16* __restrict__ brow = W + (long)(nt * 16 + l16) * K;
    v8f acc = {};
    #pragma unroll 2
    for (int kk = 0; kk < K; kk += 32) {
        v8bf a0 = *(const v8bf*)(arow + kk + g * 8);        // A: K = kk + g*8 .. +7  (VGPR 0-3)
        v8bf a1 = *(const v8bf*)(arow + kk + 16 + g * 8);   // A: K = kk+16+g*8 .. +7 (VGPR 4-7)
        v16bf av;
        #pragma unroll
        for (int i = 0; i < 8; ++i) { av[i] = a0[i]; av[8 + i] = a1[i]; }
        v8bf b0 = *(const v8bf*)(brow + kk + g * 8);        // B: K = kk + g*8 .. +7
        v8bf b1 = *(const v8bf*)(brow + kk + 16 + g * 8);   // B: K = kk+16+g*8 .. +7
        v16bf bv;
        #pragma unroll
        for (int i = 0; i < 8; ++i) { bv[i] = b0[i]; bv[8 + i] = b1[i]; }
        acc = __builtin_amdgcn_wmma_f32_16x16x32_bf16(
            false, av, false, bv, (short)0, acc, false, false);
    }
    const int ncol = nt * 16 + l16;
    #pragma unroll
    for (int v = 0; v < 8; ++v) {
        int mrow = mt * 16 + v + 8 * g;                     // f32 C/D layout
        out[(long)mrow * NG + ncol] = acc[v];
    }
}

// ---------------------------------------------------------------------------
// Per-step: LSTM elementwise.  gates (B x 4096) split [i|f|g|o].
// Writes h fp32 + bf16 copies into up to two X-buffer slots.
// grid 128 x 256 (B*1024 elements).
// ---------------------------------------------------------------------------
__global__ void lstm_elem_kernel(const float* __restrict__ gates,
                                 const float* __restrict__ bias,
                                 float* __restrict__ h, float* __restrict__ c,
                                 bf16* __restrict__ x0, int stride0,
                                 bf16* __restrict__ x1, int stride1) {
    int idx = blockIdx.x * blockDim.x + threadIdx.x;  // 0..B*ARNN-1
    int b = idx >> 10, u = idx & 1023;
    const float* g = gates + (long)b * NG;
    float gi = g[u]        + bias[u];
    float gf = g[1024 + u] + bias[1024 + u];
    float gg = g[2048 + u] + bias[2048 + u];
    float go = g[3072 + u] + bias[3072 + u];
    float cn = sigmoidf_(gf) * c[idx] + sigmoidf_(gi) * tanhf(gg);
    float hn = sigmoidf_(go) * tanhf(cn);
    c[idx] = cn; h[idx] = hn;
    bf16 hb = (bf16)hn;
    x0[(long)b * stride0 + u] = hb;
    if (x1) x1[(long)b * stride1 + u] = hb;
}

// ---------------------------------------------------------------------------
// Per-step fused location-sensitive attention. One block (256 thr) per batch.
// pq -> conv(aw,awc) -> loc-dense -> tanh energies -> masked softmax -> ctx.
// Writes aw/awc state, alignment output row, and ctx (f32 + bf16 X slots).
// ---------------------------------------------------------------------------
__global__ void attention_kernel(const float* __restrict__ ah,       // (B, ARNN)
                                 const float* __restrict__ qW,       // (ADIM, ARNN)
                                 const float* __restrict__ vW,       // (ADIM)
                                 const float* __restrict__ convW,    // (LF, 2, LK)
                                 const float* __restrict__ denseW,   // (ADIM, LF)
                                 const float* __restrict__ pmem,     // (B, T, ADIM)
                                 const float* __restrict__ memory,   // (B, T, DENC)
                                 const int*   __restrict__ memlen,   // (B)
                                 float* __restrict__ aw, float* __restrict__ awc,
                                 float* __restrict__ ctx_f32,        // (B, DENC)
                                 bf16* __restrict__ xatt_ctx,        // X_att + 256
                                 bf16* __restrict__ xdec_ctx,        // X_dec + 1024
                                 float* __restrict__ align_out) {    // d_out+ALIGN+s*T
    __shared__ float s_awp[TENC + 2 * 15];
    __shared__ float s_awcp[TENC + 2 * 15];
    __shared__ float s_pq[ADIM];
    __shared__ float s_v[ADIM];
    __shared__ float s_e[TENC];
    __shared__ float s_red[256];
    __shared__ float s_cw[LF * 2 * LK];
    __shared__ float s_dw[ADIM * LF];
    const int b = blockIdx.x, tid = threadIdx.x;
    const int len = memlen[b];

    for (int i = tid; i < LF * 2 * LK; i += 256) s_cw[i] = convW[i];
    for (int i = tid; i < ADIM * LF; i += 256)   s_dw[i] = denseW[i];
    if (tid < ADIM) s_v[tid] = vW[tid];
    for (int i = tid; i < TENC + 30; i += 256) { s_awp[i] = 0.0f; s_awcp[i] = 0.0f; }
    __syncthreads();
    for (int i = tid; i < TENC; i += 256) {
        s_awp[15 + i]  = aw[(long)b * TENC + i];
        s_awcp[15 + i] = awc[(long)b * TENC + i];
    }
    // processed query: pq[a] = query_W[a,:] . ah[b,:]
    if (tid < ADIM) {
        const float* ahb = ah + (long)b * ARNN;
        const float* qr = qW + (long)tid * ARNN;
        float acc = 0.0f;
        #pragma unroll 4
        for (int k = 0; k < ARNN; ++k) acc += qr[k] * ahb[k];
        s_pq[tid] = acc;
    }
    __syncthreads();

    // energies
    for (int t = tid; t < TENC; t += 256) {
        float loc[LF];
        #pragma unroll
        for (int f = 0; f < LF; ++f) {
            float cacc = 0.0f;
            #pragma unroll
            for (int j = 0; j < LK; ++j)
                cacc += s_cw[f * 62 + j] * s_awp[t + j] + s_cw[f * 62 + LK + j] * s_awcp[t + j];
            loc[f] = cacc;
        }
        const float* pm = pmem + ((long)b * TENC + t) * ADIM;
        float e = 0.0f;
        for (int a = 0; a < ADIM; ++a) {
            float pl = 0.0f;
            #pragma unroll
            for (int f = 0; f < LF; ++f) pl += loc[f] * s_dw[a * LF + f];
            e += s_v[a] * tanhf(s_pq[a] + pl + pm[a]);
        }
        s_e[t] = (t < len) ? e : -1e9f;
    }
    __syncthreads();

    // softmax: max
    float lm = -3.4e38f;
    for (int t = tid; t < TENC; t += 256) lm = fmaxf(lm, s_e[t]);
    s_red[tid] = lm; __syncthreads();
    for (int o = 128; o > 0; o >>= 1) {
        if (tid < o) s_red[tid] = fmaxf(s_red[tid], s_red[tid + o]);
        __syncthreads();
    }
    float mx = s_red[0]; __syncthreads();
    // exp + sum
    float ls = 0.0f;
    for (int t = tid; t < TENC; t += 256) { float ex = __expf(s_e[t] - mx); s_e[t] = ex; ls += ex; }
    s_red[tid] = ls; __syncthreads();
    for (int o = 128; o > 0; o >>= 1) {
        if (tid < o) s_red[tid] += s_red[tid + o];
        __syncthreads();
    }
    float inv = 1.0f / s_red[0];
    for (int t = tid; t < TENC; t += 256) {
        float w = s_e[t] * inv;
        s_e[t] = w;
        aw[(long)b * TENC + t]  = w;
        awc[(long)b * TENC + t] = s_awcp[15 + t] + w;
        align_out[(long)b * TOUT * TENC + t] = w;
    }
    __syncthreads();

    // context = aw . memory[b]
    for (int d = tid; d < DENC; d += 256) {
        const float* memb = memory + (long)b * TENC * DENC + d;
        float cacc = 0.0f;
        #pragma unroll 4
        for (int t = 0; t < TENC; ++t) cacc += s_e[t] * memb[(long)t * DENC];
        ctx_f32[(long)b * DENC + d] = cacc;
        bf16 cb = (bf16)cacc;
        xatt_ctx[(long)b * KATT + d] = cb;
        xdec_ctx[(long)b * KDEC + d] = cb;
    }
}

// ---------------------------------------------------------------------------
// Per-step projections: mel = [dh;ctx] @ proj_W^T + b, gate scalar.
// grid B x 128 (thread j<80 -> mel col j; j==80 -> gate).
// ---------------------------------------------------------------------------
__global__ void proj_kernel(const float* __restrict__ dh,   // (B, 1024)
                            const float* __restrict__ ctx,  // (B, 512)
                            const float* __restrict__ pW,   // (80, 1536)
                            const float* __restrict__ pb,   // (80)
                            const float* __restrict__ gW,   // (1, 1536)
                            const float* __restrict__ gb,   // (1)
                            float* __restrict__ mel_out,    // d_out
                            float* __restrict__ gate_out,   // d_out + GATE_OFF
                            int s) {
    const int b = blockIdx.x, j = threadIdx.x;
    if (j > 80) return;
    const float* W = (j < 80) ? (pW + (long)j * (ARNN + DENC)) : gW;
    const float* dhb = dh + (long)b * ARNN;
    const float* cxb = ctx + (long)b * DENC;
    float acc = 0.0f;
    #pragma unroll 4
    for (int k = 0; k < ARNN; ++k) acc += W[k] * dhb[k];
    #pragma unroll 4
    for (int k = 0; k < DENC; ++k) acc += W[ARNN + k] * cxb[k];
    if (j < 80) mel_out[(long)b * NMEL * TOUT + j * TOUT + s] = acc + pb[j];
    else        gate_out[(long)b * TOUT + s] = acc + gb[0];
}

// ---------------------------------------------------------------------------
// Host side
// ---------------------------------------------------------------------------
static inline char* carve(char*& p, size_t bytes) {
    char* r = p;
    p += (bytes + 255) & ~(size_t)255;
    return r;
}

extern "C" void kernel_launch(void* const* d_in, const int* in_sizes, int n_in,
                              void* d_out, int out_size, void* d_ws, size_t ws_size,
                              hipStream_t stream) {
    (void)in_sizes; (void)n_in; (void)out_size; (void)ws_size;
    const float* memory     = (const float*)d_in[0];
    const float* dec_inputs = (const float*)d_in[1];
    const int*   memlen     = (const int*)  d_in[2];
    const float* prenet_w1  = (const float*)d_in[3];
    const float* prenet_w2  = (const float*)d_in[4];
    const float* att_W_ih   = (const float*)d_in[5];
    const float* att_W_hh   = (const float*)d_in[6];
    const float* att_b      = (const float*)d_in[7];
    const float* query_W    = (const float*)d_in[8];
    const float* memory_W   = (const float*)d_in[9];
    const float* v_W        = (const float*)d_in[10];
    const float* loc_conv_W = (const float*)d_in[11];
    const float* loc_dense_W= (const float*)d_in[12];
    const float* dec_W_ih   = (const float*)d_in[13];
    const float* dec_W_hh   = (const float*)d_in[14];
    const float* dec_b      = (const float*)d_in[15];
    const float* proj_W     = (const float*)d_in[16];
    const float* proj_b     = (const float*)d_in[17];
    const float* gate_W     = (const float*)d_in[18];
    const float* gate_b     = (const float*)d_in[19];
    float* out = (float*)d_out;

    const long MEL_FLOATS  = (long)BQ * NMEL * TOUT;   // 1,280,000
    const long GATE_FLOATS = (long)BQ * TOUT;          //    16,000
    float* mel_out   = out;
    float* gate_out  = out + MEL_FLOATS;
    float* align_out = out + MEL_FLOATS + GATE_FLOATS;

    // workspace carve
    char* p = (char*)d_ws;
    bf16*  watt  = (bf16*) carve(p, (size_t)NG * KATT * sizeof(bf16));
    bf16*  wdec  = (bf16*) carve(p, (size_t)NG * KDEC * sizeof(bf16));
    bf16*  pren  = (bf16*) carve(p, (size_t)TOUT * BQ * PREN * sizeof(bf16));
    float* pmem  = (float*)carve(p, (size_t)BQ * TENC * ADIM * sizeof(float));
    bf16*  xatt  = (bf16*) carve(p, (size_t)BQ * KATT * sizeof(bf16));
    bf16*  xdec  = (bf16*) carve(p, (size_t)BQ * KDEC * sizeof(bf16));
    float* gates = (float*)carve(p, (size_t)BQ * NG * sizeof(float));
    // recurrent state (zeroed each launch)
    const size_t STATE_FLOATS = (size_t)BQ * (ARNN * 4 + DENC + TENC * 2);
    float* state = (float*)carve(p, STATE_FLOATS * sizeof(float));
    float* ah  = state;
    float* ac  = ah  + BQ * ARNN;
    float* dh  = ac  + BQ * ARNN;
    float* dc  = dh  + BQ * ARNN;
    float* ctx = dc  + BQ * ARNN;
    float* aw  = ctx + BQ * DENC;
    float* awc = aw  + BQ * TENC;

    // ---- setup ----
    pack_w_kernel<<<2048, 256, 0, stream>>>(att_W_ih, att_W_hh, watt, PREN + DENC, ARNN);
    pack_w_kernel<<<2048, 256, 0, stream>>>(dec_W_ih, dec_W_hh, wdec, ARNN + DENC, ARNN);
    prenet_kernel<<<dim3(TOUT, BQ), 256, 0, stream>>>(dec_inputs, prenet_w1, prenet_w2, pren);
    pmem_kernel<<<dim3(TENC, BQ), 128, 0, stream>>>(memory, memory_W, pmem);
    {
        int n = (int)STATE_FLOATS;
        zero_u32_kernel<<<(n + 255) / 256, 256, 0, stream>>>((unsigned*)state, n);
        int na = (BQ * KATT * (int)sizeof(bf16)) / 4;
        zero_u32_kernel<<<(na + 255) / 256, 256, 0, stream>>>((unsigned*)xatt, na);
        int nd = (BQ * KDEC * (int)sizeof(bf16)) / 4;
        zero_u32_kernel<<<(nd + 255) / 256, 256, 0, stream>>>((unsigned*)xdec, nd);
    }

    // ---- 500 sequential decoder steps ----
    for (int s = 0; s < TOUT; ++s) {
        // X_att[:, 0:256] = prenet frame s
        copy_pren_kernel<<<(BQ * PREN) / 256, 256, 0, stream>>>(pren + (long)s * BQ * PREN, xatt);
        // attention-LSTM gates (K=1792) via bf16 WMMA
        gemm_gates_kernel<<<64, 256, 0, stream>>>(xatt, watt, gates, KATT);
        // ah,ac update; ah -> X_att[768:] (next step) and X_dec[0:1024] (this step)
        lstm_elem_kernel<<<(BQ * ARNN) / 256, 256, 0, stream>>>(
            gates, att_b, ah, ac, xatt + (PREN + DENC), KATT, xdec, KDEC);
        // location-sensitive attention; ctx -> X_att[256:768], X_dec[1024:1536]
        attention_kernel<<<BQ, 256, 0, stream>>>(
            ah, query_W, v_W, loc_conv_W, loc_dense_W, pmem, memory, memlen,
            aw, awc, ctx, xatt + PREN, xdec + ARNN, align_out + (long)s * TENC);
        // decoder-LSTM gates (K=2560) via bf16 WMMA
        gemm_gates_kernel<<<64, 256, 0, stream>>>(xdec, wdec, gates, KDEC);
        // dh,dc update; dh -> X_dec[1536:] (next step)
        lstm_elem_kernel<<<(BQ * ARNN) / 256, 256, 0, stream>>>(
            gates, dec_b, dh, dc, xdec + (ARNN + DENC), KDEC, (bf16*)nullptr, 0);
        // mel + gate projections
        proj_kernel<<<BQ, 128, 0, stream>>>(dh, ctx, proj_W, proj_b, gate_W, gate_b,
                                            mel_out, gate_out, s);
    }
}
